// SlotAttentionExperimental_75428215652962
// MI455X (gfx1250) — matrix operand order repaired and checked by reading, not compiled, not run on hardware
//
#include <hip/hip_runtime.h>
#include <hip/hip_bf16.h>
#include <math.h>

// ---------------------------------------------------------------------------
// SlotAttention-style pipeline for MI455X (gfx1250, wave32, WMMA).
// GEMM operands live in bf16 in global memory (producers write bf16 directly;
// fp32 masters kept for state + pointwise math). GEMM tiles are staged into
// LDS with gfx1250 async-to-LDS DMA (ASYNCcnt), double-buffered, and consumed
// by v_wmma_f32_16x16x32_bf16 with fragment-native LDS layouts (ds_load_b128).
// ---------------------------------------------------------------------------

typedef __attribute__((ext_vector_type(16))) __bf16 v16bf;
typedef __attribute__((ext_vector_type(8)))  __bf16 bf16x8;
typedef __attribute__((ext_vector_type(8)))  float  v8f;

#define BATCHB 32
#define NTOK   4096
#define NSLOT  64
#define DIMD   512
#define GATE   1536
#define HIDD   2048

// GEMM tiling: block tile 128(M) x 64(N), K step 32. 8 waves / block,
// wave w owns rows [16w,16w+16), 4 WMMA 16x16 tiles across N.
#define TM 128
#define TN 64
#define TK 32
#define APITCH  40  // bf16 elems per A row in LDS (32 k + 8 pad); 80B, 16B-aligned
#define BKPITCH 40  // bf16 elems per B column in LDS (32 k + 8 pad); k-contiguous

__device__ __forceinline__ float sigmoid_f(float x) { return 1.f / (1.f + __expf(-x)); }

// --- gfx1250 async global->LDS copy of one 16B chunk (per lane) -------------
__device__ __forceinline__ void async_cp16(const __bf16* g, const __bf16* l)
{
  // GLOBAL_LOAD_ASYNC_TO_LDS_B128: vdst = LDS byte address, vaddr = 64b global.
  asm volatile("global_load_async_to_lds_b128 %0, %1, off"
               :: "v"((unsigned)(unsigned long long)l), "v"(g)
               : "memory");
}
__device__ __forceinline__ void async_wait0()
{
#if __has_builtin(__builtin_amdgcn_s_wait_asynccnt)
  __builtin_amdgcn_s_wait_asynccnt(0);
#else
  asm volatile("s_wait_asynccnt 0x0" ::: "memory");
#endif
}

// ---------------------------------------------------------------------------
// General batched GEMM:  C = alpha * A (*) op(B) + bias [+ relu]
//   A: (M,K) bf16 row-major.  transB=1: B is (N,K) bf16 (C = A*B^T), async DMA.
//                             transB=0: B is (K,N) bf16 (C = A*B), manual stage.
//   Cout: fp32 (outBf=0) or bf16 (outBf=1).
// M may be < tile (clamped loads, guarded stores); N%64==0, K%32==0 required.
// ---------------------------------------------------------------------------
__launch_bounds__(256)
__global__ void gemm_bf16_wmma(const __bf16* __restrict__ A,
                               const __bf16* __restrict__ B,
                               void* __restrict__ Cout,
                               const float* __restrict__ bias,
                               int M, int N, int K,
                               long long strA, long long strB, long long strC,
                               float alpha, int transB, int relu, int outBf)
{
  __shared__ __attribute__((aligned(16))) __bf16 As[2][TM * APITCH];
  __shared__ __attribute__((aligned(16))) __bf16 Bs[2][TN * BKPITCH];

  const long long bz = blockIdx.z;
  A += bz * strA; B += bz * strB;

  const int m0   = blockIdx.y * TM;
  const int n0   = blockIdx.x * TN;
  const int tid  = threadIdx.x;
  const int wave = tid >> 5;
  const int lane = tid & 31;

  v8f acc[4] = {};

  // WMMA wave32 fragment addressing (CDNA5 ISA 7.12.2):
  const int aRow = lane & 15;
  const int aKb0 = (lane < 16) ? 0 : 8;
  const int aKb1 = (lane < 16) ? 16 : 24;
  const int bCol = lane & 15;
  const int bKb  = (lane < 16) ? 0 : 8;

  // ---- async tile staging (16B chunk per lane) ----
  auto stageA_async = [&](int kk, int bufi) {
#pragma unroll
    for (int p = 0; p < 2; ++p) {
      int q = p * 256 + tid;               // 512 chunks over TM x TK
      int r = q >> 2, c8 = (q & 3) * 8;
      int gr = m0 + r; if (gr >= M) gr = M - 1;
      async_cp16(A + (long long)gr * K + (kk + c8), &As[bufi][r * APITCH + c8]);
    }
  };
  auto stageB_async = [&](int kk, int bufi) {
    int n = tid >> 2, c8 = (tid & 3) * 8;  // 256 chunks over TN x TK
    async_cp16(B + (long long)(n0 + n) * K + (kk + c8),
               &Bs[bufi][n * BKPITCH + c8]);
  };
  bf16x8 pBn;
  auto fetchB_nn = [&](int kk) {           // B (K,N): chunk along n
    int k = tid >> 3, n8 = (tid & 7) * 8;
    pBn = *(const bf16x8*)(B + (long long)(kk + k) * N + (n0 + n8));
  };
  auto storeB_nn = [&](int bufi) {         // transpose scatter into Bs[n][k]
    int k = tid >> 3, n8 = (tid & 7) * 8;
#pragma unroll
    for (int i = 0; i < 8; ++i) Bs[bufi][(n8 + i) * BKPITCH + k] = pBn[i];
  };
  auto compute = [&](int bufi) {
    const __bf16* arow = &As[bufi][(wave * 16 + aRow) * APITCH];
    bf16x8 a0 = *(const bf16x8*)(arow + aKb0);
    bf16x8 a1 = *(const bf16x8*)(arow + aKb1);
    v16bf afrag = __builtin_shufflevector(a0, a1, 0, 1, 2, 3, 4, 5, 6, 7,
                                          8, 9, 10, 11, 12, 13, 14, 15);
#pragma unroll
    for (int t = 0; t < 4; ++t) {
      const __bf16* bb = &Bs[bufi][(t * 16 + bCol) * BKPITCH + bKb];
      bf16x8 b0 = *(const bf16x8*)(bb);
      bf16x8 b1 = *(const bf16x8*)(bb + 16);
      v16bf bfrag = __builtin_shufflevector(b0, b1, 0, 1, 2, 3, 4, 5, 6, 7,
                                            8, 9, 10, 11, 12, 13, 14, 15);
      acc[t] = __builtin_amdgcn_wmma_f32_16x16x32_bf16(
          false, afrag, false, bfrag, (short)0, acc[t], false, false);
    }
  };

  // ---- prologue: stage tile 0 into buffer 0 ----
  stageA_async(0, 0);
  if (transB) stageB_async(0, 0);
  else { fetchB_nn(0); storeB_nn(0); }
  async_wait0();
  __syncthreads();

  // ---- main loop: double-buffered LDS, async prefetch of next tile ----
  int buf = 0;
  for (int k0 = 0; k0 < K; k0 += TK) {
    const bool hasNext = (k0 + TK) < K;
    if (hasNext) {
      stageA_async(k0 + TK, buf ^ 1);
      if (transB) stageB_async(k0 + TK, buf ^ 1);
      else fetchB_nn(k0 + TK);
    }
    compute(buf);
    if (hasNext && !transB) storeB_nn(buf ^ 1);
    if (hasNext) async_wait0();
    __syncthreads();
    buf ^= 1;
  }

  // ---- epilogue: C/D layout: VGPR r -> rows r (lanes 0-15) / r+8 (16-31) ----
#pragma unroll
  for (int t = 0; t < 4; ++t) {
    int col = n0 + t * 16 + bCol;
    float bv = bias ? bias[col] : 0.f;
#pragma unroll
    for (int r = 0; r < 8; ++r) {
      int grow = m0 + wave * 16 + r + ((lane < 16) ? 0 : 8);
      if (grow < M) {
        float v = acc[t][r] * alpha + bv;
        if (relu) v = fmaxf(v, 0.f);
        if (outBf) ((__bf16*)Cout + bz * strC)[(long long)grow * N + col] = (__bf16)v;
        else       ((float*)Cout  + bz * strC)[(long long)grow * N + col] = v;
      }
    }
  }
}

// ---------------------------------------------------------------------------
__global__ void cvt_f32_bf16_kernel(const float* __restrict__ in,
                                    __bf16* __restrict__ out, long long n)
{
  long long i = (long long)blockIdx.x * blockDim.x + threadIdx.x;
  if (i < n) out[i] = (__bf16)in[i];
}

// slots = mu + sigma * noise; write fp32 master + bf16 mirror
__global__ void slots_init_kernel(const float* __restrict__ mu,
                                  const float* __restrict__ sg,
                                  const float* __restrict__ noise,
                                  float* __restrict__ s32,
                                  __bf16* __restrict__ s16, long long n)
{
  long long i = (long long)blockIdx.x * blockDim.x + threadIdx.x;
  if (i >= n) return;
  int d = (int)(i & (DIMD - 1));
  float v = mu[d] + sg[d] * noise[i];
  s32[i] = v; s16[i] = (__bf16)v;
}

// ---------------------------------------------------------------------------
// LayerNorm over rows of 512 (fp32 in -> bf16 out); one wave per row.
// ---------------------------------------------------------------------------
__launch_bounds__(256)
__global__ void layernorm512_kernel(const float* __restrict__ x,
                                    __bf16* __restrict__ y,
                                    const float* __restrict__ g,
                                    const float* __restrict__ b,
                                    long long rows)
{
  long long row = (long long)blockIdx.x * 8 + (threadIdx.x >> 5);
  if (row >= rows) return;
  int lane = threadIdx.x & 31;
  const float4* xr = (const float4*)(x + row * DIMD);
  float4 v[4]; float s = 0.f, ss = 0.f;
#pragma unroll
  for (int i = 0; i < 4; ++i) {
    v[i] = xr[lane + i * 32];
    s  += v[i].x + v[i].y + v[i].z + v[i].w;
    ss += v[i].x * v[i].x + v[i].y * v[i].y + v[i].z * v[i].z + v[i].w * v[i].w;
  }
#pragma unroll
  for (int o = 16; o > 0; o >>= 1) { s += __shfl_xor(s, o, 32); ss += __shfl_xor(ss, o, 32); }
  float mean = s * (1.f / DIMD);
  float var  = ss * (1.f / DIMD) - mean * mean;
  float inv  = rsqrtf(var + 1e-5f);
  __bf16* yr = y + row * DIMD;
  const float4* gr = (const float4*)g;
  const float4* br = (const float4*)b;
#pragma unroll
  for (int i = 0; i < 4; ++i) {
    int idx = lane + i * 32;
    float4 gg = gr[idx], bb = br[idx];
    __bf16* o4 = yr + idx * 4;
    o4[0] = (__bf16)((v[i].x - mean) * inv * gg.x + bb.x);
    o4[1] = (__bf16)((v[i].y - mean) * inv * gg.y + bb.y);
    o4[2] = (__bf16)((v[i].z - mean) * inv * gg.z + bb.z);
    o4[3] = (__bf16)((v[i].w - mean) * inv * gg.w + bb.w);
  }
}

// ---------------------------------------------------------------------------
// a1 attention: dots (32, 64, 4096) fp32. softmax over slots (axis 1) + EPS.
// ---------------------------------------------------------------------------
__global__ void softmax_slots_col_kernel(float* __restrict__ dots)
{
  int t = blockIdx.x * blockDim.x + threadIdx.x;   // b * 4096
  if (t >= BATCHB * NTOK) return;
  int b = t >> 12, j = t & (NTOK - 1);
  float* p = dots + (long long)b * NSLOT * NTOK + j;
  float m = -INFINITY;
#pragma unroll 4
  for (int i = 0; i < NSLOT; ++i) m = fmaxf(m, p[(long long)i * NTOK]);
  float s = 0.f;
#pragma unroll 4
  for (int i = 0; i < NSLOT; ++i) {
    float e = __expf(p[(long long)i * NTOK] - m);
    p[(long long)i * NTOK] = e; s += e;
  }
  float inv = 1.f / s;
#pragma unroll 4
  for (int i = 0; i < NSLOT; ++i)
    p[(long long)i * NTOK] = p[(long long)i * NTOK] * inv + 1e-8f;
}

// a1: divide each (b,i) row of 4096 by its sum, write bf16 attn.
__launch_bounds__(256)
__global__ void rownorm4096_kernel(const float* __restrict__ dots,
                                   __bf16* __restrict__ attn)
{
  long long row = blockIdx.x;           // b * 64 rows
  const float* p = dots + row * NTOK;
  __bf16* o = attn + row * NTOK;
  __shared__ float red[8];
  float s = 0.f;
  for (int i = threadIdx.x; i < NTOK; i += 256) s += p[i];
#pragma unroll
  for (int o2 = 16; o2 > 0; o2 >>= 1) s += __shfl_xor(s, o2, 32);
  if ((threadIdx.x & 31) == 0) red[threadIdx.x >> 5] = s;
  __syncthreads();
  if (threadIdx.x == 0) {
    float t = 0.f;
#pragma unroll
    for (int w = 0; w < 8; ++w) t += red[w];
    red[0] = 1.f / t;
  }
  __syncthreads();
  float inv = red[0];
  for (int i = threadIdx.x; i < NTOK; i += 256) o[i] = (__bf16)(p[i] * inv);
}

// ---------------------------------------------------------------------------
// a2 attention: dots (32, 4096, 64) fp32. softmax over last dim + EPS.
// ---------------------------------------------------------------------------
__launch_bounds__(256)
__global__ void softmax_row64_kernel(float* __restrict__ dots, long long rows)
{
  long long row = (long long)blockIdx.x * 8 + (threadIdx.x >> 5);
  if (row >= rows) return;
  int lane = threadIdx.x & 31;
  float* p = dots + row * NSLOT;
  float a = p[lane], b = p[lane + 32];
  float m = fmaxf(a, b);
#pragma unroll
  for (int o = 16; o > 0; o >>= 1) m = fmaxf(m, __shfl_xor(m, o, 32));
  a = __expf(a - m); b = __expf(b - m);
  float s = a + b;
#pragma unroll
  for (int o = 16; o > 0; o >>= 1) s += __shfl_xor(s, o, 32);
  float inv = 1.f / s;
  p[lane]      = a * inv + 1e-8f;
  p[lane + 32] = b * inv + 1e-8f;
}

// a2: reciprocal column sums over axis 1 (4096 rows) -> cs[b*64+j]
__global__ void colsum_kernel(const float* __restrict__ dots, float* __restrict__ cs)
{
  int t = blockIdx.x * blockDim.x + threadIdx.x;
  if (t >= BATCHB * NSLOT) return;
  int b = t >> 6, j = t & (NSLOT - 1);
  const float* p = dots + (long long)b * NTOK * NSLOT + j;
  float s = 0.f;
  for (int i = 0; i < NTOK; ++i) s += p[(long long)i * NSLOT];
  cs[t] = 1.f / s;
}

__global__ void coldiv_kernel(const float* __restrict__ dots,
                              const float* __restrict__ cs,
                              __bf16* __restrict__ attn)
{
  long long t = (long long)blockIdx.x * blockDim.x + threadIdx.x;
  if (t >= (long long)BATCHB * NTOK * NSLOT) return;
  int j = (int)(t & (NSLOT - 1));
  long long b = t >> 18;                     // / (4096*64)
  attn[t] = (__bf16)(dots[t] * cs[b * NSLOT + j]);
}

// ---------------------------------------------------------------------------
// GRU combine: gi/gh (M,1536) fp32 pre-activations (bias fused in GEMM),
// h (M,512) fp32 master. Writes fp32 master + bf16 mirror. out may alias h.
// ---------------------------------------------------------------------------
__global__ void gru_combine_kernel(const float* __restrict__ gi,
                                   const float* __restrict__ gh,
                                   const float* __restrict__ h,
                                   float* __restrict__ out,
                                   __bf16* __restrict__ outbf, long long M)
{
  long long i = (long long)blockIdx.x * blockDim.x + threadIdx.x;
  if (i >= M * DIMD) return;
  long long m = i >> 9; int d = (int)(i & (DIMD - 1));
  const float* gim = gi + m * GATE;
  const float* ghm = gh + m * GATE;
  float r = sigmoid_f(gim[d] + ghm[d]);
  float z = sigmoid_f(gim[DIMD + d] + ghm[DIMD + d]);
  float n = tanhf(gim[2 * DIMD + d] + r * ghm[2 * DIMD + d]);
  float v = (1.f - z) * n + z * h[i];
  out[i] = v; outbf[i] = (__bf16)v;
}

// ---------------------------------------------------------------------------
// Host orchestration.  Input flattening (setup_inputs dict insertion order):
//  0 inputs, 1 slot_noise, 2 slots_mu, 3 slots_sigma, 4 ni_g, 5 ni_b,
//  a1: 6 ng 7 nb 8 cg 9 cb 10 Wq 11 Wk 12 Wv 13 Wih 14 Whh 15 bih 16 bhh
//  f1: 17 ng 18 nb 19 W1 20 b1 21 W2 22 b2 23 Wih 24 Whh 25 bih 26 bhh
//  a2: 27..37 (as a1), f2: 38..47 (as f1)
// ---------------------------------------------------------------------------
extern "C" void kernel_launch(void* const* d_in, const int* in_sizes, int n_in,
                              void* d_out, int out_size, void* d_ws, size_t ws_size,
                              hipStream_t stream)
{
  (void)in_sizes; (void)n_in; (void)out_size; (void)ws_size;

  const float* P[48];
  for (int i = 0; i < 48; ++i) P[i] = (const float*)d_in[i];
  const float* inputs = P[0];
  const float* noise  = P[1];
  const float* mu = P[2]; const float* sg = P[3];
  const float* nig = P[4]; const float* nib = P[5];

  struct AttnP { const float *ng,*nb,*cg,*cb,*Wq,*Wk,*Wv,*Wih,*Whh,*bih,*bhh; };
  struct FfP   { const float *ng,*nb,*W1,*b1,*W2,*b2,*Wih,*Whh,*bih,*bhh; };
  AttnP a1{P[6],P[7],P[8],P[9],P[10],P[11],P[12],P[13],P[14],P[15],P[16]};
  FfP   f1{P[17],P[18],P[19],P[20],P[21],P[22],P[23],P[24],P[25],P[26]};
  AttnP a2{P[27],P[28],P[29],P[30],P[31],P[32],P[33],P[34],P[35],P[36],P[37]};
  FfP   f2{P[38],P[39],P[40],P[41],P[42],P[43],P[44],P[45],P[46],P[47]};

  const long long NX   = (long long)BATCHB * NTOK  * DIMD;   // 67,108,864
  const long long NS   = (long long)BATCHB * NSLOT * DIMD;   //  1,048,576
  const long long NDOT = (long long)BATCHB * NSLOT * NTOK;   //  8,388,608
  const long long MX   = (long long)BATCHB * NTOK;           // 131,072 rows
  const long long MS   = (long long)BATCHB * NSLOT;          //   2,048 rows
  const int CH = 16384;                                       // x-path chunk rows
  const int NCHUNK = (int)(MX / CH);
  const long long WQn = (long long)DIMD * DIMD;               // 262,144
  const long long WGn = (long long)GATE * DIMD;               // 786,432
  const long long W1n = (long long)HIDD * DIMD;               // 1,048,576
  const long long W2n = (long long)DIMD * HIDD;               // 1,048,576

  // ---- workspace layout: fp32 region then bf16 region ----
  float* wf = (float*)d_ws;
  float* x_f   = wf; wf += NX;
  float* sl_f  = wf; wf += NS;
  float* dots  = wf; wf += NDOT;
  float* csum  = wf; wf += 2048;
  float* gi    = wf; wf += (long long)CH * GATE;
  float* gh    = wf; wf += (long long)CH * GATE;

  __bf16* wb = (__bf16*)wf;
  __bf16* x_bf  = wb; wb += NX;    // bf16 mirror of x
  __bf16* lnx   = wb; wb += NX;    // LN output (x-sized)
  __bf16* bbuf1 = wb; wb += NX;    // k(a1) / q(a2)
  __bf16* bbuf2 = wb; wb += NX;    // v(a1) / updates(a2)
  __bf16* sl_bf = wb; wb += NS;    // bf16 mirror of slots
  __bf16* lns   = wb; wb += NS;
  __bf16* sbuf1 = wb; wb += NS;    // q(a1) / k(a2)
  __bf16* sbuf2 = wb; wb += NS;    // v(a2)
  __bf16* sbuf3 = wb; wb += NS;    // updates(a1) / y(f1)
  __bf16* attn  = wb; wb += NDOT;
  __bf16* ffh   = wb; wb += (long long)CH * HIDD;
  __bf16* ybf   = wb; wb += (long long)CH * DIMD;
  // bf16 weights
  __bf16 *a1Wq = wb; wb += WQn; __bf16 *a1Wk = wb; wb += WQn; __bf16 *a1Wv = wb; wb += WQn;
  __bf16 *a1Wih = wb; wb += WGn; __bf16 *a1Whh = wb; wb += WGn;
  __bf16 *f1W1 = wb; wb += W1n; __bf16 *f1W2 = wb; wb += W2n;
  __bf16 *f1Wih = wb; wb += WGn; __bf16 *f1Whh = wb; wb += WGn;
  __bf16 *a2Wq = wb; wb += WQn; __bf16 *a2Wk = wb; wb += WQn; __bf16 *a2Wv = wb; wb += WQn;
  __bf16 *a2Wih = wb; wb += WGn; __bf16 *a2Whh = wb; wb += WGn;
  __bf16 *f2W1 = wb; wb += W1n; __bf16 *f2W2 = wb; wb += W2n;
  __bf16 *f2Wih = wb; wb += WGn; __bf16 *f2Whh = wb; wb += WGn;

  const float scale = 1.f / sqrtf((float)DIMD);

  auto cvt = [&](const float* in, __bf16* out, long long n) {
    cvt_f32_bf16_kernel<<<dim3((unsigned)((n + 255) / 256)), dim3(256), 0, stream>>>(in, out, n);
  };
  auto gemm = [&](const __bf16* A, const __bf16* B, void* C, const float* bias,
                  int M, int N, int K, long long sA, long long sB, long long sC,
                  int batch, float alpha, int transB, int relu, int outBf) {
    dim3 g(N / TN, (M + TM - 1) / TM, batch);
    gemm_bf16_wmma<<<g, dim3(256), 0, stream>>>(A, B, C, bias, M, N, K,
                                                sA, sB, sC, alpha, transB, relu, outBf);
  };
  auto ln = [&](const float* in, __bf16* out, const float* g, const float* b,
                long long rows) {
    layernorm512_kernel<<<dim3((unsigned)((rows + 7) / 8)), dim3(256), 0, stream>>>(
        in, out, g, b, rows);
  };
  auto gru = [&](const float* gi_, const float* gh_, const float* h,
                 float* out, __bf16* outbf, long long M) {
    long long n = M * DIMD;
    gru_combine_kernel<<<dim3((unsigned)((n + 255) / 256)), dim3(256), 0, stream>>>(
        gi_, gh_, h, out, outbf, M);
  };

  // ---- convert weights fp32 -> bf16 (every call; deterministic) ----
  cvt(a1.Wq, a1Wq, WQn); cvt(a1.Wk, a1Wk, WQn); cvt(a1.Wv, a1Wv, WQn);
  cvt(a1.Wih, a1Wih, WGn); cvt(a1.Whh, a1Whh, WGn);
  cvt(f1.W1, f1W1, W1n); cvt(f1.W2, f1W2, W2n);
  cvt(f1.Wih, f1Wih, WGn); cvt(f1.Whh, f1Whh, WGn);
  cvt(a2.Wq, a2Wq, WQn); cvt(a2.Wk, a2Wk, WQn); cvt(a2.Wv, a2Wv, WQn);
  cvt(a2.Wih, a2Wih, WGn); cvt(a2.Whh, a2Whh, WGn);
  cvt(f2.W1, f2W1, W1n); cvt(f2.W2, f2W2, W2n);
  cvt(f2.Wih, f2Wih, WGn); cvt(f2.Whh, f2Whh, WGn);

  // ---- init: slots = mu + sigma*noise; x = LN(inputs) (bf16 + fp32 master) --
  slots_init_kernel<<<dim3((unsigned)((NS + 255) / 256)), dim3(256), 0, stream>>>(
      mu, sg, noise, sl_f, sl_bf, NS);
  ln(inputs, x_bf, nig, nib, MX);                 // bf16 mirror of initial x
  cvt_f32_bf16_kernel<<<1, 1, 0, stream>>>(nullptr, nullptr, 0); // no-op spacer
  // fp32 master of initial x: recompute LN into x_f via a tiny trick — reuse
  // gru-free path: x_f = LN(inputs) in fp32 is needed; do it with a dedicated
  // conversion from bf16 would lose bits, so run LN again writing fp32 via
  // bf16->fp32 upconvert of the bf16 mirror (state precision is bf16-bound
  // anyway for GEMM consumers; masters only feed LN/GRU pointwise):
  // Simpler: keep x_f as upconverted mirror.
  {
    long long n = NX;
    // upconvert bf16 -> fp32
    struct L { static __global__ void k(const __bf16* a, float* b, long long n); };
    // (defined below as a real kernel)
    extern __global__ void cvt_bf16_f32_kernel(const __bf16*, float*, long long);
    cvt_bf16_f32_kernel<<<dim3((unsigned)((n + 255) / 256)), dim3(256), 0, stream>>>(
        x_bf, x_f, n);
  }

  for (int it = 0; it < 3; ++it) {
    // ================= a1: slots attend over x (softmax over slots) =========
    ln(sl_f, lns, a1.ng, a1.nb, MS);
    ln(x_f,  lnx, a1.cg, a1.cb, MX);
    gemm(lns, a1Wq, sbuf1, nullptr, (int)MS, DIMD, DIMD, 0, 0, 0, 1, 1.f, 1, 0, 1); // q
    gemm(lnx, a1Wk, bbuf1, nullptr, (int)MX, DIMD, DIMD, 0, 0, 0, 1, 1.f, 1, 0, 1); // k
    gemm(lnx, a1Wv, bbuf2, nullptr, (int)MX, DIMD, DIMD, 0, 0, 0, 1, 1.f, 1, 0, 1); // v
    gemm(sbuf1, bbuf1, dots, nullptr, NSLOT, NTOK, DIMD,
         (long long)NSLOT * DIMD, (long long)NTOK * DIMD, (long long)NSLOT * NTOK,
         BATCHB, scale, 1, 0, 0);                                   // dots fp32
    softmax_slots_col_kernel<<<dim3((BATCHB * NTOK) / 256), dim3(256), 0, stream>>>(dots);
    rownorm4096_kernel<<<dim3((unsigned)(BATCHB * NSLOT)), dim3(256), 0, stream>>>(dots, attn);
    gemm(attn, bbuf2, sbuf3, nullptr, NSLOT, DIMD, NTOK,
         (long long)NSLOT * NTOK, (long long)NTOK * DIMD, (long long)NSLOT * DIMD,
         BATCHB, 1.f, 0, 0, 1);                                     // updates bf16
    gemm(sbuf3, a1Wih, gi, a1.bih, (int)MS, GATE, DIMD, 0, 0, 0, 1, 1.f, 1, 0, 0);
    gemm(sl_bf, a1Whh, gh, a1.bhh, (int)MS, GATE, DIMD, 0, 0, 0, 1, 1.f, 1, 0, 0);
    gru(gi, gh, sl_f, sl_f, sl_bf, MS);

    // ================= f1: FF on slots + GRU ================================
    ln(sl_f, lns, f1.ng, f1.nb, MS);
    gemm(lns, f1W1, ffh,   f1.b1, (int)MS, HIDD, DIMD, 0, 0, 0, 1, 1.f, 1, 1, 1);
    gemm(ffh, f1W2, sbuf3, f1.b2, (int)MS, DIMD, HIDD, 0, 0, 0, 1, 1.f, 1, 0, 1);
    gemm(sbuf3, f1Wih, gi, f1.bih, (int)MS, GATE, DIMD, 0, 0, 0, 1, 1.f, 1, 0, 0);
    gemm(sl_bf, f1Whh, gh, f1.bhh, (int)MS, GATE, DIMD, 0, 0, 0, 1, 1.f, 1, 0, 0);
    gru(gi, gh, sl_f, sl_f, sl_bf, MS);

    // ================= a2: x attends over slots (softmax over slots) ========
    ln(x_f,  lnx, a2.ng, a2.nb, MX);
    ln(sl_f, lns, a2.cg, a2.cb, MS);
    gemm(lnx, a2Wq, bbuf1, nullptr, (int)MX, DIMD, DIMD, 0, 0, 0, 1, 1.f, 1, 0, 1); // q
    gemm(lns, a2Wk, sbuf1, nullptr, (int)MS, DIMD, DIMD, 0, 0, 0, 1, 1.f, 1, 0, 1); // k
    gemm(lns, a2Wv, sbuf2, nullptr, (int)MS, DIMD, DIMD, 0, 0, 0, 1, 1.f, 1, 0, 1); // v
    gemm(bbuf1, sbuf1, dots, nullptr, NTOK, NSLOT, DIMD,
         (long long)NTOK * DIMD, (long long)NSLOT * DIMD, (long long)NTOK * NSLOT,
         BATCHB, scale, 1, 0, 0);                                   // dots fp32
    softmax_row64_kernel<<<dim3((unsigned)((MX + 7) / 8)), dim3(256), 0, stream>>>(dots, MX);
    colsum_kernel<<<dim3((BATCHB * NSLOT + 255) / 256), dim3(256), 0, stream>>>(dots, csum);
    coldiv_kernel<<<dim3((unsigned)((NDOT + 255) / 256)), dim3(256), 0, stream>>>(dots, csum, attn);
    gemm(attn, sbuf2, bbuf2, nullptr, NTOK, DIMD, NSLOT,
         (long long)NTOK * NSLOT, (long long)NSLOT * DIMD, (long long)NTOK * DIMD,
         BATCHB, 1.f, 0, 0, 1);                                     // updates bf16
    for (int c = 0; c < NCHUNK; ++c) {
      long long r0 = (long long)c * CH;
      gemm(bbuf2 + r0 * DIMD, a2Wih, gi, a2.bih, CH, GATE, DIMD, 0, 0, 0, 1, 1.f, 1, 0, 0);
      gemm(x_bf  + r0 * DIMD, a2Whh, gh, a2.bhh, CH, GATE, DIMD, 0, 0, 0, 1, 1.f, 1, 0, 0);
      gru(gi, gh, x_f + r0 * DIMD, x_f + r0 * DIMD, x_bf + r0 * DIMD, CH);
    }

    // ================= f2: FF on x + GRU (chunked) ==========================
    ln(x_f, lnx, f2.ng, f2.nb, MX);
    for (int c = 0; c < NCHUNK; ++c) {
      long long r0 = (long long)c * CH;
      gemm(lnx + r0 * DIMD, f2W1, ffh, f2.b1, CH, HIDD, DIMD, 0, 0, 0, 1, 1.f, 1, 1, 1);
      gemm(ffh, f2W2, ybf, f2.b2, CH, DIMD, HIDD, 0, 0, 0, 1, 1.f, 1, 0, 1);
      gemm(ybf, f2Wih, gi, f2.bih, CH, GATE, DIMD, 0, 0, 0, 1, 1.f, 1, 0, 0);
      gemm(x_bf + r0 * DIMD, f2Whh, gh, f2.bhh, CH, GATE, DIMD, 0, 0, 0, 1, 1.f, 1, 0, 0);
      gru(gi, gh, x_f + r0 * DIMD, x_f + r0 * DIMD, x_bf + r0 * DIMD, CH);
    }
  }

  // ---- output: [slots | x] concatenated flat (fp32 masters) ----
  hipMemcpyAsync(d_out, sl_f, NS * sizeof(float), hipMemcpyDeviceToDevice, stream);
  hipMemcpyAsync((float*)d_out + NS, x_f, NX * sizeof(float),
                 hipMemcpyDeviceToDevice, stream);
}

// bf16 -> fp32 upconvert (defined after use; declared via extern above)
__global__ void cvt_bf16_f32_kernel(const __bf16* __restrict__ in,
                                    float* __restrict__ out, long long n)
{
  long long i = (long long)blockIdx.x * blockDim.x + threadIdx.x;
  if (i < n) out[i] = (float)in[i];
}
